// ASCNet_54279796687163
// MI455X (gfx1250) — matrix operand order repaired and checked by reading, not compiled, not run on hardware
//
#include <hip/hip_runtime.h>
#include <hip/hip_bf16.h>

#define NTOT 131072   // B*H*W = 8*128*128
#define HW   16384    // H*W

typedef __attribute__((ext_vector_type(16))) __bf16 bf16x16;
typedef __attribute__((ext_vector_type(8)))  float  f32x8;
typedef __attribute__((ext_vector_type(4)))  unsigned int u32x4;
typedef __attribute__((ext_vector_type(8)))  int  i32x8;
typedef __attribute__((ext_vector_type(4)))  int  i32x4;

// ---------- bf16 helpers (bit-exact RNE, no reliance on __bf16 arithmetic) ----------
static __device__ __forceinline__ __bf16 tobf(float f) {
  unsigned u = __float_as_uint(f);
  unsigned r = (u + 0x7fffu + ((u >> 16) & 1u)) >> 16;
  unsigned short s = (unsigned short)r;
  return __builtin_bit_cast(__bf16, s);
}
static __device__ __forceinline__ float tof(__bf16 h) {
  unsigned short s = __builtin_bit_cast(unsigned short, h);
  return __uint_as_float(((unsigned)s) << 16);
}

// ---------- WMMA fragment loaders (CDNA5 16-bit layouts, wave32) ----------
// Fragments are fetched as two 16-byte chunks -> ds_load_b128 / global_load_b128.
static __device__ __forceinline__ bf16x16 load_a_frag(const __bf16* A, int row0, int lda, int k0) {
  int lane = threadIdx.x & 31;
  const __bf16* p = A + (size_t)(row0 + (lane & 15)) * lda + k0 + ((lane >> 4) ? 8 : 0);
  union { uint4 q[2]; bf16x16 v; } u;
  u.q[0] = *(const uint4*)p;         // K +0..7
  u.q[1] = *(const uint4*)(p + 16);  // K +16..23
  return u.v;
}
static __device__ __forceinline__ bf16x16 load_b_frag_t(const __bf16* Bt, int col0, int ldk, int k0) {
  int lane = threadIdx.x & 31;
  const __bf16* p = Bt + (size_t)(col0 + (lane & 15)) * ldk + k0 + ((lane >> 4) ? 8 : 0);
  union { uint4 q[2]; bf16x16 v; } u;
  u.q[0] = *(const uint4*)p;
  u.q[1] = *(const uint4*)(p + 16);
  return u.v;
}
static __device__ __forceinline__ f32x8 wmma_bf16(f32x8 acc, bf16x16 a, bf16x16 b) {
  return __builtin_amdgcn_wmma_f32_16x16x32_bf16(false, a, false, b, (short)0, acc, false, false);
}

// ---------- Tensor Data Mover: 2D tile (rows x cols of 2-byte elems) global -> LDS ----------
#if __has_builtin(__builtin_amdgcn_tensor_load_to_lds)
#define HAVE_TDM 1
static __device__ __forceinline__ void tdm_load_2d(unsigned lds_addr, const void* gptr,
                                                   unsigned rows, unsigned cols,
                                                   unsigned tensor_w, unsigned tensor_h,
                                                   unsigned stride_elems) {
  unsigned long long ga = (unsigned long long)(size_t)gptr;
  u32x4 g0;
  g0[0] = 1u;                                     // count=1, user mode, no gather
  g0[1] = lds_addr;                               // LDS byte address
  g0[2] = (unsigned)(ga & 0xffffffffu);           // global_addr[31:0]
  g0[3] = (unsigned)((ga >> 32) & 0x01ffffffu)    // global_addr[56:32]
        | (2u << 30);                             // type = 2 ("image")
  i32x8 g1;
  g1[0] = (int)(1u << 16);                        // wg_mask=0, data_size=1 (2 bytes)
  g1[1] = (int)((tensor_w & 0xffffu) << 16);      // atomic_addr=0 | tensor_dim0 lo16
  g1[2] = (int)(((tensor_w >> 16) & 0xffffu) | ((tensor_h & 0xffffu) << 16));
  g1[3] = (int)(((tensor_h >> 16) & 0xffffu) | ((cols & 0xffffu) << 16));  // tile_dim0
  g1[4] = (int)(rows & 0xffffu);                  // tile_dim1 | tile_dim2=0
  g1[5] = (int)stride_elems;                      // tensor_dim0_stride lo32
  g1[6] = 0;                                      // stride hi16 | dim1_stride lo16
  g1[7] = 0;
  i32x4 z4 = {0, 0, 0, 0};
  i32x8 z8 = {0, 0, 0, 0, 0, 0, 0, 0};
  // 6-arg form (clang-23 / therock headers): g0, g1, g2, g3, extra, cpol
  __builtin_amdgcn_tensor_load_to_lds(g0, g1, z4, z4, z8, 0);
}
#else
#define HAVE_TDM 0
#endif

// ---------- weight prep ----------
__global__ void k_cvt(const float* __restrict__ s, __bf16* __restrict__ d, int n) {
  int i = blockIdx.x * 256 + threadIdx.x;
  if (i < n) d[i] = tobf(s[i]);
}
__global__ void k_catgb(const float* gq, const float* bq, const float* gk, const float* bk,
                        const float* gv, const float* bv, float* g512, float* b512) {
  int i = blockIdx.x * 256 + threadIdx.x;
  if (i >= 512) return;
  float g, b;
  if (i < 128)      { g = gq[i];       b = bq[i]; }
  else if (i < 256) { g = gk[i - 128]; b = bk[i - 128]; }
  else              { g = gv[i - 256]; b = bv[i - 256]; }
  g512[i] = g; b512[i] = b;
}

// ---------- conv3x3 (64->128, pad 1) + BN + relu as implicit-GEMM WMMA ----------
__global__ void __launch_bounds__(256) k_conv3(const float* __restrict__ x,
                                               const __bf16* __restrict__ A0,
                                               const float* __restrict__ g,
                                               const float* __restrict__ bta,
                                               __bf16* __restrict__ out) {
  __shared__ __bf16 LX[64 * 3 * 132];      // input window [cin][kh][x], x = w'+1, w' in [-1,128]
  __shared__ unsigned short koff[576];     // k -> LDS element offset (add w)
  __shared__ __bf16 Bt[128 * 40];          // K-chunk in WMMA layout [col][k 0..31 (+pad)]
  int bid = blockIdx.x;
  int bb = bid >> 7, h = bid & 127;
  int tid = threadIdx.x;
  for (int k = tid; k < 576; k += 256) {
    int cin = k / 9, r = k % 9, kh = r / 3, kw = r % 3;
    koff[k] = (unsigned short)((cin * 3 + kh) * 132 + kw);
  }
  for (int idx = tid; idx < 64 * 3 * 130; idx += 256) {
    int cin = idx / 390, r2 = idx % 390, kh = r2 / 130, xx = r2 % 130;
    int wp = xx - 1, hp = h + kh - 1;
    float v = 0.f;
    if (wp >= 0 && wp < 128 && hp >= 0 && hp < 128)
      v = x[(((size_t)bb * 64 + cin) * 128 + hp) * 128 + wp];
    LX[(cin * 3 + kh) * 132 + xx] = tobf(v);
  }
  int wave = tid >> 5, lane = tid & 31;
  int kr = tid >> 3, col0 = (tid & 7) * 16;
  f32x8 acc[8];
#pragma unroll
  for (int s = 0; s < 8; ++s)
#pragma unroll
    for (int r = 0; r < 8; ++r) acc[s][r] = 0.f;
  for (int k0 = 0; k0 < 576; k0 += 32) {
    __syncthreads();
    {  // stage im2col chunk: row k0+kr, cols col0..col0+15
      int off = koff[k0 + kr];
#pragma unroll
      for (int i = 0; i < 16; ++i) Bt[(col0 + i) * 40 + kr] = LX[off + col0 + i];
    }
    __syncthreads();
    bf16x16 af = load_a_frag(A0, wave * 16, 576, k0);
#pragma unroll
    for (int s = 0; s < 8; ++s) {
      bf16x16 bf = load_b_frag_t(Bt, s * 16, 40, 0);
      acc[s] = wmma_bf16(acc[s], af, bf);
    }
  }
  int rbase = wave * 16 + ((lane >> 4) ? 8 : 0);
#pragma unroll
  for (int s = 0; s < 8; ++s) {
    int col = s * 16 + (lane & 15);
#pragma unroll
    for (int r = 0; r < 8; ++r) {
      int row = rbase + r;
      float v = fmaxf(g[row] * acc[s][r] + bta[row], 0.f);
      out[(size_t)row * NTOT + (size_t)bb * HW + h * 128 + col] = tobf(v);
    }
  }
}

// ---------- cf = relu(bn(conv1x1(cb, wen))) ----------
__global__ void k_cf(const __bf16* __restrict__ CB, const float* __restrict__ wen,
                     const float* __restrict__ gen, const float* __restrict__ ben,
                     float* __restrict__ CF) {
  int idx = blockIdx.x * 256 + threadIdx.x;
  int kch = idx >> 17, n = idx & (NTOT - 1);
  float s = 0.f;
  for (int c = 0; c < 128; ++c) s += wen[kch * 128 + c] * tof(CB[(size_t)c * NTOT + n]);
  CF[(size_t)kch * NTOT + n] = fmaxf(gen[kch] * s + ben[kch], 0.f);
}

// ---------- CCAM coefficients: e -> softmax(max - e) ----------
__global__ void __launch_bounds__(256) k_ccam(const __bf16* __restrict__ CB,
                                              const float* __restrict__ CF,
                                              float* __restrict__ acoef) {
  __shared__ float red[256 * 8];
  __shared__ float e8[8];
  int c = blockIdx.x, bb = blockIdx.y, tid = threadIdx.x;
  float p[8];
#pragma unroll
  for (int k = 0; k < 8; ++k) p[k] = 0.f;
  for (int n = tid; n < HW; n += 256) {
    float cv = tof(CB[(size_t)c * NTOT + bb * HW + n]);
#pragma unroll
    for (int k = 0; k < 8; ++k) p[k] += cv * CF[(size_t)k * NTOT + bb * HW + n];
  }
#pragma unroll
  for (int k = 0; k < 8; ++k) red[tid * 8 + k] = p[k];
  __syncthreads();
  if (tid < 8) {
    float s = 0.f;
    for (int t = 0; t < 256; ++t) s += red[t * 8 + tid];
    e8[tid] = s;
  }
  __syncthreads();
  if (tid == 0) {
    float mn = e8[0];
    for (int k = 1; k < 8; ++k) mn = fminf(mn, e8[k]);
    float w[8], sum = 0.f;
    for (int k = 0; k < 8; ++k) { w[k] = __expf(mn - e8[k]); sum += w[k]; }
    float inv = 1.f / sum;
    for (int k = 0; k < 8; ++k) acoef[((size_t)bb * 128 + c) * 8 + k] = w[k] * inv;
  }
}

// ---------- xs = cb + sc*dec (in-place on CB) ----------
__global__ void k_xs(__bf16* __restrict__ CB, const float* __restrict__ CF,
                     const float* __restrict__ acoef, const float* __restrict__ sc) {
  int c = blockIdx.y;
  int n = blockIdx.x * 256 + threadIdx.x;
  int bb = n >> 14;
  const float* a = &acoef[((size_t)bb * 128 + c) * 8];
  float dec = 0.f;
#pragma unroll
  for (int k = 0; k < 8; ++k) dec += a[k] * CF[(size_t)k * NTOT + n];
  size_t o = (size_t)c * NTOT + n;
  CB[o] = tobf(tof(CB[o]) + sc[0] * dec);
}

// ---------- generic WMMA GEMM with TDM double-buffered B staging ----------
// out = affine(A[MxK] * B[KxN]); epi=1 fuses h-sigmoid gate * qkv -> d_out NCHW
__global__ void __launch_bounds__(256) k_gemm(const __bf16* __restrict__ A,
                                              const __bf16* __restrict__ B,
                                              const float* __restrict__ g,
                                              const float* __restrict__ bta,
                                              __bf16* __restrict__ outb,
                                              const __bf16* __restrict__ gate,
                                              float* __restrict__ outf,
                                              int M, int K, int N, int epi) {
  __shared__ __align__(16) __bf16 Raw[2][32 * 128];  // row-major DMA landing buffers
  __shared__ __bf16 Bt[128 * 40];                    // WMMA layout [col][k (+pad)]
  int tid = threadIdx.x, wave = tid >> 5, lane = tid & 31;
  int n0 = blockIdx.x * 128, m0 = blockIdx.y * 128;
  f32x8 acc[8];
#pragma unroll
  for (int s = 0; s < 8; ++s)
#pragma unroll
    for (int r = 0; r < 8; ++r) acc[s][r] = 0.f;
  int kr = tid >> 3, cb0 = (tid & 7) * 16;
#if HAVE_TDM
  if (wave == 0)  // prologue: DMA tile 0
    tdm_load_2d((unsigned)(size_t)&Raw[0][0], B + (size_t)n0, 32, 128,
                (unsigned)N, (unsigned)K, (unsigned)N);
  for (int k0 = 0; k0 < K; k0 += 32) {
    int b = (k0 >> 5) & 1;
    __syncthreads();  // Bt consumed, Raw[b^1] free
    if (wave == 0) {
      if (k0 + 32 < K) {  // issue next tile, then wait for current (<=1 outstanding)
        tdm_load_2d((unsigned)(size_t)&Raw[b ^ 1][0], B + (size_t)(k0 + 32) * N + n0,
                    32, 128, (unsigned)N, (unsigned)K, (unsigned)N);
        __builtin_amdgcn_s_wait_tensorcnt(1);
      } else {
        __builtin_amdgcn_s_wait_tensorcnt(0);
      }
    }
    __syncthreads();  // Raw[b] valid for everyone
    {  // LDS->LDS transpose into WMMA layout
      union { uint4 q[2]; __bf16 e[16]; } u;
      const __bf16* rp = &Raw[b][kr * 128 + cb0];
      u.q[0] = *(const uint4*)rp;
      u.q[1] = *(const uint4*)(rp + 8);
#pragma unroll
      for (int i = 0; i < 16; ++i) Bt[(cb0 + i) * 40 + kr] = u.e[i];
    }
    __syncthreads();
    bf16x16 af = load_a_frag(A, m0 + wave * 16, K, k0);
#pragma unroll
    for (int s = 0; s < 8; ++s) {
      bf16x16 bf = load_b_frag_t(Bt, s * 16, 40, 0);
      acc[s] = wmma_bf16(acc[s], af, bf);
    }
  }
#else
  for (int k0 = 0; k0 < K; k0 += 32) {
    __syncthreads();
    const __bf16* src = B + (size_t)(k0 + kr) * N + n0 + cb0;
    if (k0 + 32 < K) __builtin_prefetch((const void*)(src + (size_t)32 * N), 0, 0);
    {
      union { uint4 q[2]; __bf16 e[16]; } u;
      u.q[0] = *(const uint4*)src;
      u.q[1] = *(const uint4*)(src + 8);
#pragma unroll
      for (int i = 0; i < 16; ++i) Bt[(cb0 + i) * 40 + kr] = u.e[i];
    }
    __syncthreads();
    bf16x16 af = load_a_frag(A, m0 + wave * 16, K, k0);
#pragma unroll
    for (int s = 0; s < 8; ++s) {
      bf16x16 bf = load_b_frag_t(Bt, s * 16, 40, 0);
      acc[s] = wmma_bf16(acc[s], af, bf);
    }
  }
#endif
  int rbase = m0 + wave * 16 + ((lane >> 4) ? 8 : 0);
#pragma unroll
  for (int s = 0; s < 8; ++s) {
    int col = n0 + s * 16 + (lane & 15);
#pragma unroll
    for (int r = 0; r < 8; ++r) {
      int row = rbase + r;
      float v = g[row] * acc[s][r] + bta[row];
      if (epi == 0) {
        outb[(size_t)row * N + col] = tobf(v);
      } else {  // final: h_sigmoid(v) * qkv -> d_out NCHW
        float att = fminf(fmaxf(v + 3.f, 0.f), 6.f) * (1.f / 6.f);
        float res = att * tof(gate[(size_t)row * N + col]);
        size_t o = (size_t)(col >> 14) * (128 * HW) + (size_t)row * HW + (col & (HW - 1));
        outf[o] = res;
      }
    }
  }
}

// ---------- depthwise 3x3 (512 groups) + BN + relu ----------
__global__ void k_dw(const __bf16* __restrict__ Q, const float* __restrict__ wd,
                     const float* __restrict__ g, const float* __restrict__ bta,
                     __bf16* __restrict__ out) {
  int idx = blockIdx.x * 256 + threadIdx.x;
  int ch = idx >> 17;
  int n = idx & (NTOT - 1);
  int bb = n >> 14, hw = n & (HW - 1), h = hw >> 7, w = hw & 127;
  float s = 0.f;
#pragma unroll
  for (int kh = 0; kh < 3; ++kh) {
    int hp = h + kh - 1;
    if (hp < 0 || hp > 127) continue;
#pragma unroll
    for (int kw = 0; kw < 3; ++kw) {
      int wp = w + kw - 1;
      if (wp < 0 || wp > 127) continue;
      s += wd[ch * 9 + kh * 3 + kw] * tof(Q[(size_t)ch * NTOT + bb * HW + hp * 128 + wp]);
    }
  }
  out[(size_t)ch * NTOT + n] = tobf(fmaxf(g[ch] * s + bta[ch], 0.f));
}

// ---------- shunt pooling (row: mean over w%4==j ; col: mean over h-block j) ----------
__global__ void k_shunt(const __bf16* __restrict__ src, int chShift, int mode,
                        float* __restrict__ out) {
  int idx = blockIdx.x * 256 + threadIdx.x;
  int l = idx & 127;
  int ch = (idx >> 7) & ((1 << chShift) - 1);
  int rest = idx >> (7 + chShift);
  int j = rest & 3, bb = rest >> 2;
  float s = 0.f;
  if (mode == 0) {
    for (int wb = 0; wb < 32; ++wb)
      s += tof(src[(size_t)ch * NTOT + bb * HW + l * 128 + wb * 4 + j]);
  } else {
    for (int hb = 0; hb < 32; ++hb)
      s += tof(src[(size_t)ch * NTOT + bb * HW + (j * 32 + hb) * 128 + l]);
  }
  out[idx] = s * (1.f / 32.f);
}

// ---------- linear resize of pos embedding (16 -> 128, half-pixel, edge clamp) ----------
static __device__ __forceinline__ float pinterp(const float* p, int l) {
  float s = (l + 0.5f) * 0.125f - 0.5f;
  float fl = floorf(s);
  int i0 = (int)fl;
  float t = s - fl;
  int a = i0 < 0 ? 0 : (i0 > 15 ? 15 : i0);
  int b = (i0 + 1) < 0 ? 0 : ((i0 + 1) > 15 ? 15 : i0 + 1);
  return p[a] * (1.f - t) + p[b] * t;
}

// ---------- axial attention per (b, chunk, head): softmax(QK^T*0.25) V, relu ----------
__global__ void __launch_bounds__(256) k_attn(const float* __restrict__ SQ,
                                              const float* __restrict__ SK,
                                              const float* __restrict__ SV,
                                              const float* __restrict__ pq,
                                              const float* __restrict__ pk,
                                              __bf16* __restrict__ outR) {
  __shared__ __align__(16) char smem[60416];
  __bf16* qh  = (__bf16*)smem;              // [128][32], K zero-padded 16..31
  __bf16* khT = (__bf16*)(smem + 8192);     // [128 cols(L)][32 K], padded
  __bf16* vT  = (__bf16*)(smem + 16384);    // [32 d][136 L]
  __bf16* Sb  = (__bf16*)(smem + 25088);    // [128][136] logits -> probs (in place)
  int bid = blockIdx.x;
  int bb = bid >> 5, j = (bid >> 3) & 3, head = bid & 7;
  int tid = threadIdx.x;
  for (int idx = tid; idx < 128 * 32; idx += 256) {
    int l = idx >> 5, k = idx & 31;
    float qv = 0.f, kv = 0.f;
    if (k < 16) {
      int ch = head * 16 + k;
      size_t o = ((size_t)(bb * 4 + j) * 128 + ch) * 128 + l;
      qv = SQ[o] + pinterp(pq + ((size_t)j * 128 + ch) * 16, l);
      kv = SK[o] + pinterp(pk + ((size_t)j * 128 + ch) * 16, l);
    }
    qh[l * 32 + k]  = tobf(qv);
    khT[l * 32 + k] = tobf(kv);
  }
  for (int idx = tid; idx < 32 * 128; idx += 256) {
    int d = idx >> 7, l = idx & 127;
    vT[d * 136 + l] = tobf(SV[((size_t)(bb * 4 + j) * 256 + head * 32 + d) * 128 + l]);
  }
  __syncthreads();
  int wave = tid >> 5, lane = tid & 31;
  {  // S = Q K^T * SCALE
    bf16x16 af = load_a_frag(qh, wave * 16, 32, 0);
    int rbase = wave * 16 + ((lane >> 4) ? 8 : 0);
#pragma unroll
    for (int s = 0; s < 8; ++s) {
      f32x8 a;
#pragma unroll
      for (int r = 0; r < 8; ++r) a[r] = 0.f;
      bf16x16 bf = load_b_frag_t(khT, s * 16, 32, 0);
      a = wmma_bf16(a, af, bf);
      int col = s * 16 + (lane & 15);
#pragma unroll
      for (int r = 0; r < 8; ++r) Sb[(rbase + r) * 136 + col] = tobf(a[r] * 0.25f);
    }
  }
  __syncthreads();
  if (tid < 128) {  // softmax rows, in place
    __bf16* row = Sb + tid * 136;
    float m = -1e30f;
    for (int k = 0; k < 128; ++k) m = fmaxf(m, tof(row[k]));
    float s = 0.f;
    for (int k = 0; k < 128; ++k) s += __expf(tof(row[k]) - m);
    float inv = 1.f / s;
    for (int k = 0; k < 128; ++k) row[k] = tobf(__expf(tof(row[k]) - m) * inv);
  }
  __syncthreads();
  {  // O = P V  (M=128, N=32, K=128)
    f32x8 acc0, acc1;
#pragma unroll
    for (int r = 0; r < 8; ++r) { acc0[r] = 0.f; acc1[r] = 0.f; }
    for (int k0 = 0; k0 < 128; k0 += 32) {
      bf16x16 af = load_a_frag(Sb, wave * 16, 136, k0);
      bf16x16 b0 = load_b_frag_t(vT, 0, 136, k0);
      bf16x16 b1 = load_b_frag_t(vT, 16, 136, k0);
      acc0 = wmma_bf16(acc0, af, b0);
      acc1 = wmma_bf16(acc1, af, b1);
    }
    int rbase = wave * 16 + ((lane >> 4) ? 8 : 0);
#pragma unroll
    for (int r = 0; r < 8; ++r) {
      int l = rbase + r;
      int d0 = (lane & 15), d1 = 16 + (lane & 15);
      size_t n = (size_t)bb * 512 + j * 128 + l;
      outR[((size_t)(head * 32 + d0)) * 4096 + n] = tobf(fmaxf(acc0[r], 0.f));
      outR[((size_t)(head * 32 + d1)) * 4096 + n] = tobf(fmaxf(acc1[r], 0.f));
    }
  }
}

// ---------- xx = broadcast(xr) + broadcast(xc) + v, relu ----------
__global__ void k_xx(const __bf16* __restrict__ XRC, const __bf16* __restrict__ XCC,
                     const __bf16* __restrict__ V, __bf16* __restrict__ XXR) {
  int idx = blockIdx.x * 256 + threadIdx.x;
  int ch = idx >> 17, n = idx & (NTOT - 1);
  int bb = n >> 14, p = n & (HW - 1);
  int gq = p >> 10, r = (p >> 5) & 31, c = p & 31;
  float xr = tof(XRC[(size_t)ch * 4096 + bb * 512 + gq * 32 + r]);
  float xc = tof(XCC[(size_t)ch * 4096 + bb * 512 + gq * 32 + c]);
  float vv = tof(V[(size_t)ch * NTOT + n]);
  XXR[(size_t)ch * NTOT + n] = tobf(fmaxf(xr + xc + vv, 0.f));
}

// =====================================================================================
extern "C" void kernel_launch(void* const* d_in, const int* in_sizes, int n_in,
                              void* d_out, int out_size, void* d_ws, size_t ws_size,
                              hipStream_t stream) {
  (void)in_sizes; (void)n_in; (void)out_size; (void)ws_size;
  const float* x   = (const float*)d_in[0];
  const float* wcc = (const float*)d_in[1];
  const float* gcc = (const float*)d_in[2];
  const float* bcc = (const float*)d_in[3];
  const float* wen = (const float*)d_in[4];
  const float* gen = (const float*)d_in[5];
  const float* ben = (const float*)d_in[6];
  const float* sc  = (const float*)d_in[7];
  const float* wq  = (const float*)d_in[8];
  const float* gq  = (const float*)d_in[9];
  const float* bq  = (const float*)d_in[10];
  const float* wk  = (const float*)d_in[11];
  const float* gk  = (const float*)d_in[12];
  const float* bk  = (const float*)d_in[13];
  const float* wv  = (const float*)d_in[14];
  const float* gv  = (const float*)d_in[15];
  const float* bv  = (const float*)d_in[16];
  const float* wd  = (const float*)d_in[17];
  const float* gd  = (const float*)d_in[18];
  const float* bd  = (const float*)d_in[19];
  const float* wp  = (const float*)d_in[20];
  const float* gp  = (const float*)d_in[21];
  const float* bp  = (const float*)d_in[22];
  const float* wr  = (const float*)d_in[23];
  const float* gr  = (const float*)d_in[24];
  const float* br  = (const float*)d_in[25];
  const float* wc2 = (const float*)d_in[26];
  const float* gc2 = (const float*)d_in[27];
  const float* bc2 = (const float*)d_in[28];
  const float* wo  = (const float*)d_in[29];
  const float* go  = (const float*)d_in[30];
  const float* bo  = (const float*)d_in[31];
  const float* prq = (const float*)d_in[32];
  const float* prk = (const float*)d_in[33];
  const float* pcq = (const float*)d_in[34];
  const float* pck = (const float*)d_in[35];

  char* ws = (char*)d_ws;
  size_t off = 0;
  auto carve = [&](size_t bytes) -> char* {
    char* p = ws + off;
    off = (off + bytes + 255) & ~(size_t)255;
    return p;
  };
  // weights (bf16) — total ~433 MB workspace
  __bf16* A0   = (__bf16*)carve((size_t)128 * 576 * 2);
  __bf16* Aqkv = (__bf16*)carve((size_t)512 * 128 * 2);
  __bf16* Ap   = (__bf16*)carve((size_t)128 * 512 * 2);
  __bf16* Ar   = (__bf16*)carve((size_t)256 * 256 * 2);
  __bf16* Ac2  = (__bf16*)carve((size_t)256 * 256 * 2);
  __bf16* Ao   = (__bf16*)carve((size_t)128 * 256 * 2);
  float* g512  = (float*)carve(512 * 4);
  float* b512  = (float*)carve(512 * 4);
  // activations, channel-major [C][B*HW]
  __bf16* CB   = (__bf16*)carve((size_t)128 * NTOT * 2);  // cb, then xs in place
  float*  CF   = (float*)carve((size_t)8 * NTOT * 4);
  float*  ACO  = (float*)carve((size_t)8 * 128 * 8 * 4);
  __bf16* QKV  = (__bf16*)carve((size_t)512 * NTOT * 2);  // [q;k;v]
  __bf16* DW   = (__bf16*)carve((size_t)512 * NTOT * 2);
  __bf16* QKVP = (__bf16*)carve((size_t)128 * NTOT * 2);
  float* SQr = (float*)carve((size_t)8 * 4 * 128 * 128 * 4);
  float* SKr = (float*)carve((size_t)8 * 4 * 128 * 128 * 4);
  float* SVr = (float*)carve((size_t)8 * 4 * 256 * 128 * 4);
  float* SQc = (float*)carve((size_t)8 * 4 * 128 * 128 * 4);
  float* SKc = (float*)carve((size_t)8 * 4 * 128 * 128 * 4);
  float* SVc = (float*)carve((size_t)8 * 4 * 256 * 128 * 4);
  __bf16* XRR = (__bf16*)carve((size_t)256 * 4096 * 2);   // relu(attn out) row
  __bf16* XCR = (__bf16*)carve((size_t)256 * 4096 * 2);
  __bf16* XRC = (__bf16*)carve((size_t)256 * 4096 * 2);   // after wr conv
  __bf16* XCC = (__bf16*)carve((size_t)256 * 4096 * 2);
  __bf16* XXR = (__bf16*)carve((size_t)256 * NTOT * 2);
  float* OUT = (float*)d_out;

  // weight prep
  k_cvt<<<288, 256, 0, stream>>>(wcc, A0, 73728);
  k_cvt<<<64, 256, 0, stream>>>(wq, Aqkv, 16384);
  k_cvt<<<64, 256, 0, stream>>>(wk, Aqkv + 16384, 16384);
  k_cvt<<<128, 256, 0, stream>>>(wv, Aqkv + 32768, 32768);
  k_cvt<<<256, 256, 0, stream>>>(wp, Ap, 65536);
  k_cvt<<<256, 256, 0, stream>>>(wr, Ar, 65536);
  k_cvt<<<256, 256, 0, stream>>>(wc2, Ac2, 65536);
  k_cvt<<<128, 256, 0, stream>>>(wo, Ao, 32768);
  k_catgb<<<2, 256, 0, stream>>>(gq, bq, gk, bk, gv, bv, g512, b512);

  // pipeline
  k_conv3<<<1024, 256, 0, stream>>>(x, A0, gcc, bcc, CB);
  k_cf<<<4096, 256, 0, stream>>>(CB, wen, gen, ben, CF);
  k_ccam<<<dim3(128, 8), 256, 0, stream>>>(CB, CF, ACO);
  k_xs<<<dim3(512, 128), 256, 0, stream>>>(CB, CF, ACO, sc);
  // q/k/v in one GEMM: [512x128] x [128 x 131072]
  k_gemm<<<dim3(1024, 4), 256, 0, stream>>>(Aqkv, CB, g512, b512, QKV, nullptr, nullptr,
                                            512, 128, NTOT, 0);
  k_dw<<<262144, 256, 0, stream>>>(QKV, wd, gd, bd, DW);
  k_gemm<<<dim3(1024, 1), 256, 0, stream>>>(Ap, DW, gp, bp, QKVP, nullptr, nullptr,
                                            128, 512, NTOT, 0);
  // shunts (q at +0, k at +128ch, v at +256ch)
  k_shunt<<<2048, 256, 0, stream>>>(QKV, 7, 0, SQr);
  k_shunt<<<2048, 256, 0, stream>>>(QKV + (size_t)128 * NTOT, 7, 0, SKr);
  k_shunt<<<4096, 256, 0, stream>>>(QKV + (size_t)256 * NTOT, 8, 0, SVr);
  k_shunt<<<2048, 256, 0, stream>>>(QKV, 7, 1, SQc);
  k_shunt<<<2048, 256, 0, stream>>>(QKV + (size_t)128 * NTOT, 7, 1, SKc);
  k_shunt<<<4096, 256, 0, stream>>>(QKV + (size_t)256 * NTOT, 8, 1, SVc);
  // axial attention (row + column)
  k_attn<<<256, 256, 0, stream>>>(SQr, SKr, SVr, prq, prk, XRR);
  k_attn<<<256, 256, 0, stream>>>(SQc, SKc, SVc, pcq, pck, XCR);
  // wr / wc2 1x1 convs
  k_gemm<<<dim3(32, 2), 256, 0, stream>>>(Ar, XRR, gr, br, XRC, nullptr, nullptr,
                                          256, 256, 4096, 0);
  k_gemm<<<dim3(32, 2), 256, 0, stream>>>(Ac2, XCR, gc2, bc2, XCC, nullptr, nullptr,
                                          256, 256, 4096, 0);
  // xx = xr + xc + v, relu
  k_xx<<<131072, 256, 0, stream>>>(XRC, XCC, QKV + (size_t)256 * NTOT, XXR);
  // final: att = h_sigmoid(bn(wo * xx)); out = att * qkv  -> d_out (NCHW fp32)
  k_gemm<<<dim3(1024, 1), 256, 0, stream>>>(Ao, XXR, go, bo, nullptr, QKVP, OUT,
                                            128, 256, NTOT, 1);
}
